// CrossAttention_24378234372776
// MI455X (gfx1250) — matrix operand order repaired and checked by reading, not compiled, not run on hardware
//
#include <hip/hip_runtime.h>
#include <hip/hip_bf16.h>
#include <math.h>

// ---------------------------------------------------------------------------
// CrossAttention on MI455X (gfx1250): flash-attention fused pipeline, all
// matmuls on v_wmma_f32_16x16x32_f16 (f16 in, f32 accumulate).
//   b=4, heads=8, dim_head=64, queries=4096 (64x64), keys=1024 (32x32)
// Weights are pre-converted to f16 once (ws), so projection inner loops are
// pure {global_load_b128 -> v_wmma} with no per-tile cvt work.
// ---------------------------------------------------------------------------

#define BATCH 4
#define HEADS 8
#define DHEAD 64
#define HID   512
#define CIN   256
#define NQ    4096
#define NK    1024

typedef _Float16 h16;
typedef __attribute__((ext_vector_type(16))) _Float16 v16h;
typedef __attribute__((ext_vector_type(8)))  _Float16 v8h;
typedef __attribute__((ext_vector_type(8)))  float    v8f;

#define WMMA_F16(A, B, C) \
  __builtin_amdgcn_wmma_f32_16x16x32_f16(false, (A), false, (B), (short)0, (C), false, false)

__device__ __forceinline__ v16h cat8(v8h a, v8h b) {
  return __builtin_shufflevector(a, b, 0,1,2,3,4,5,6,7,8,9,10,11,12,13,14,15);
}

// ---------------------------------------------------------------------------
// Kernel 0: one-shot fp32 -> f16 weight conversion (grid-stride).
// ---------------------------------------------------------------------------
__global__ __launch_bounds__(256) void wcvt_kernel(const float* __restrict__ src,
                                                   h16* __restrict__ dst, int n) {
  for (int i = blockIdx.x * 256 + threadIdx.x; i < n; i += gridDim.x * 256)
    dst[i] = (h16)src[i];
}

// ---------------------------------------------------------------------------
// Kernel 1: Q projection.  q[b,h,i,d] = 0.125 * sum_c x[b,c,i] * Wq[h*64+d,c]
// Block: 64 pixels x 512 outs, 256 threads (8 waves). x tile staged in LDS.
// ---------------------------------------------------------------------------
__global__ __launch_bounds__(256) void qproj_kernel(const float* __restrict__ x,
                                                    const h16* __restrict__ Wq16,
                                                    h16* __restrict__ Qo) {
  __shared__ h16 xs[64][CIN + 8];
  const int b  = blockIdx.y;
  const int pt = blockIdx.x;                 // 64-pixel tile, 0..63
  const int tid = threadIdx.x;

  const float* xb = x + ((size_t)b * CIN) * NQ + (size_t)pt * 64;
#pragma unroll 4
  for (int it = 0; it < 64; ++it) {          // 16384 f32, coalesced per channel
    int idx = it * 256 + tid;
    int c = idx >> 6, p = idx & 63;
    xs[p][c] = (h16)xb[(size_t)c * NQ + p];
  }
  __syncthreads();

  const int wv = tid >> 5, lane = tid & 31;
  const int lane15 = lane & 15, hi = lane >> 4;
  const int base8 = hi * 8, base16 = hi * 16;
  const int mrow = (wv & 3) * 16 + lane15;   // A-matrix row (pixel)

  v16h Aq[8];                                 // K=256 -> 8 chained WMMAs
#pragma unroll
  for (int kc = 0; kc < 8; ++kc)
    Aq[kc] = cat8(*(const v8h*)&xs[mrow][kc * 32 + base16],
                  *(const v8h*)&xs[mrow][kc * 32 + base16 + 8]);

  const int nbase = (wv >> 2) * 256;
  for (int nt = 0; nt < 16; ++nt) {
    const int o = nbase + nt * 16 + lane15;  // out channel (B column per lane)
    const h16* wrow = Wq16 + (size_t)o * CIN + base16;
    v8f acc = {};
#pragma unroll
    for (int kc = 0; kc < 8; ++kc)
      acc = WMMA_F16(Aq[kc], *(const v16h*)(wrow + kc * 32), acc);

    const int head = o >> 6, d = o & 63;
    h16* qdst = Qo + (((size_t)b * HEADS + head) * NQ) * DHEAD + d;
#pragma unroll
    for (int r = 0; r < 8; ++r) {
      int qi = pt * 64 + (wv & 3) * 16 + r + 8 * hi;
      qdst[(size_t)qi * DHEAD] = (h16)(acc[r] * 0.125f);   // 64^-0.5 folded in
    }
  }
}

// ---------------------------------------------------------------------------
// Kernel 2: KV projection. half=0 -> K[b,h,j,d], half=1 -> V^T[b,h,d,j]
// (V transposed so the PV B-matrix load is contiguous in the key dimension.)
// ---------------------------------------------------------------------------
__global__ __launch_bounds__(256) void kvproj_kernel(const float* __restrict__ a,
                                                     const h16* __restrict__ Wkv16,
                                                     h16* __restrict__ Ko,
                                                     h16* __restrict__ VTo) {
  __shared__ h16 as_[64][CIN + 8];
  const int b    = blockIdx.z;
  const int half = blockIdx.y;               // 0 = K weights, 1 = V weights
  const int pt   = blockIdx.x;               // 0..15 (1024/64)
  const int tid  = threadIdx.x;

  const float* ab = a + ((size_t)b * CIN) * NK + (size_t)pt * 64;
#pragma unroll 4
  for (int it = 0; it < 64; ++it) {
    int idx = it * 256 + tid;
    int c = idx >> 6, p = idx & 63;
    as_[p][c] = (h16)ab[(size_t)c * NK + p];
  }
  __syncthreads();

  const int wv = tid >> 5, lane = tid & 31;
  const int lane15 = lane & 15, hi = lane >> 4;
  const int base8 = hi * 8, base16 = hi * 16;
  const int mrow = (wv & 3) * 16 + lane15;

  v16h Aq[8];
#pragma unroll
  for (int kc = 0; kc < 8; ++kc)
    Aq[kc] = cat8(*(const v8h*)&as_[mrow][kc * 32 + base16],
                  *(const v8h*)&as_[mrow][kc * 32 + base16 + 8]);

  const int nbase = (wv >> 2) * 256;
  for (int nt = 0; nt < 16; ++nt) {
    const int o = nbase + nt * 16 + lane15;  // channel within the 512 half
    const h16* wrow = Wkv16 + (size_t)(half * HID + o) * CIN + base16;
    v8f acc = {};
#pragma unroll
    for (int kc = 0; kc < 8; ++kc)
      acc = WMMA_F16(Aq[kc], *(const v16h*)(wrow + kc * 32), acc);

    const int head = o >> 6, d = o & 63;
    if (half == 0) {
      h16* kd = Ko + (((size_t)b * HEADS + head) * NK) * DHEAD + d;
#pragma unroll
      for (int r = 0; r < 8; ++r) {
        int j = pt * 64 + (wv & 3) * 16 + r + 8 * hi;
        kd[(size_t)j * DHEAD] = (h16)acc[r];
      }
    } else {
      h16* vd = VTo + (((size_t)b * HEADS + head) * DHEAD + d) * NK;
#pragma unroll
      for (int r = 0; r < 8; ++r) {
        int j = pt * 64 + (wv & 3) * 16 + r + 8 * hi;
        vd[j] = (h16)acc[r];
      }
    }
  }
}

// ---------------------------------------------------------------------------
// Kernel 3: flash attention.  One wave per 16-query tile of one (b, head).
// Per 32-key block: S = Q*K^T (4 WMMA), online softmax (shfl_xor reductions
// across the 16-lane half that owns a C-row), P staged via wave-private LDS
// (C-layout -> A-layout transpose), O += P*V (4 WMMA). 256 WMMA per wave.
// ---------------------------------------------------------------------------
__global__ __launch_bounds__(128) void flash_kernel(const h16* __restrict__ Q,
                                                    const h16* __restrict__ Kx,
                                                    const h16* __restrict__ VT,
                                                    h16* __restrict__ AO) {
  __shared__ h16 pst[4][16 * 40];            // per-wave P staging, padded rows
  const int tid = threadIdx.x;
  const int wv = tid >> 5, lane = tid & 31;
  const int lane15 = lane & 15, hi = lane >> 4;
  const int base8 = hi * 8, base16 = hi * 16;

  const int gid = blockIdx.x * 4 + wv;       // 8192 waves total
  const int b  = gid >> 11;                  // / (8*256)
  const int h  = (gid >> 8) & 7;
  const int qt = gid & 255;                  // 16-query tile
  h16* lp = &pst[wv][0];

  // Q A-tiles (16x64 -> two 16x32 A operands), already scaled
  const h16* qp = Q + (((size_t)b * HEADS + h) * NQ + qt * 16 + lane15) * DHEAD;
  v16h Aq0 = cat8(*(const v8h*)(qp + base16),      *(const v8h*)(qp + 8 + base16));
  v16h Aq1 = cat8(*(const v8h*)(qp + 32 + base16), *(const v8h*)(qp + 40 + base16));

  float m_[8], l_[8];
  v8f O0 = {}, O1 = {}, O2 = {}, O3 = {};
#pragma unroll
  for (int r = 0; r < 8; ++r) { m_[r] = -3.0e38f; l_[r] = 0.0f; }

  const h16* kbase = Kx + (((size_t)b * HEADS + h) * NK) * DHEAD;
  const h16* vbase = VT + (((size_t)b * HEADS + h) * DHEAD) * NK;

  for (int jb = 0; jb < NK; jb += 32) {
    // ---- S = Q * K^T for keys [jb, jb+32) : two 16x16 N-tiles -------------
    const h16* kp0 = kbase + (size_t)(jb + lane15) * DHEAD;  // B column = key row
    const h16* kp1 = kp0 + 16 * DHEAD;
    // prefetch next key block's K rows (global_prefetch_b8)
    if (jb + 32 < NK) {
      __builtin_prefetch(kp0 + 32 * DHEAD, 0, 0);
      __builtin_prefetch(kp1 + 32 * DHEAD, 0, 0);
    }
    v8f S0 = {}, S1 = {};
    S0 = WMMA_F16(Aq0, *(const v16h*)(kp0 + base16),      S0);
    S0 = WMMA_F16(Aq1, *(const v16h*)(kp0 + 32 + base16), S0);
    S1 = WMMA_F16(Aq0, *(const v16h*)(kp1 + base16),      S1);
    S1 = WMMA_F16(Aq1, *(const v16h*)(kp1 + 32 + base16), S1);

    // ---- online softmax (rows r+8*hi live across one 16-lane half) --------
#pragma unroll
    for (int r = 0; r < 8; ++r) {
      float mx = fmaxf(S0[r], S1[r]);
      mx = fmaxf(mx, __shfl_xor(mx, 1));
      mx = fmaxf(mx, __shfl_xor(mx, 2));
      mx = fmaxf(mx, __shfl_xor(mx, 4));
      mx = fmaxf(mx, __shfl_xor(mx, 8));
      float mnew  = fmaxf(m_[r], mx);
      float alpha = __expf(m_[r] - mnew);
      float p0 = __expf(S0[r] - mnew);
      float p1 = __expf(S1[r] - mnew);
      float rs = p0 + p1;
      rs += __shfl_xor(rs, 1);
      rs += __shfl_xor(rs, 2);
      rs += __shfl_xor(rs, 4);
      rs += __shfl_xor(rs, 8);
      l_[r] = l_[r] * alpha + rs;
      m_[r] = mnew;
      O0[r] *= alpha; O1[r] *= alpha; O2[r] *= alpha; O3[r] *= alpha;
      // stage P (C-layout -> LDS); wave-private region, DS ops in-order
      lp[(r + 8 * hi) * 40 + lane15]      = (h16)p0;
      lp[(r + 8 * hi) * 40 + 16 + lane15] = (h16)p1;
    }

    // ---- reload P as a 16x32 A operand (two 16B DS loads per lane) --------
    v16h Ap = cat8(*(const v8h*)(lp + lane15 * 40 + base16),
                   *(const v8h*)(lp + lane15 * 40 + 8 + base16));

    // ---- O += P * V : V^T rows are contiguous in the key dim --------------
    O0 = WMMA_F16(Ap, *(const v16h*)(vbase + (size_t)(0 * 16 + lane15) * NK + jb + base16), O0);
    O1 = WMMA_F16(Ap, *(const v16h*)(vbase + (size_t)(1 * 16 + lane15) * NK + jb + base16), O1);
    O2 = WMMA_F16(Ap, *(const v16h*)(vbase + (size_t)(2 * 16 + lane15) * NK + jb + base16), O2);
    O3 = WMMA_F16(Ap, *(const v16h*)(vbase + (size_t)(3 * 16 + lane15) * NK + jb + base16), O3);
  }

  // ---- normalize and store attn-out as [b, i, head*64+d] (f16) ------------
  h16* ao = AO + ((size_t)b * NQ) * HID + h * DHEAD + lane15;
#pragma unroll
  for (int r = 0; r < 8; ++r) {
    float inv = 1.0f / l_[r];
    size_t row = (size_t)(qt * 16 + r + 8 * hi) * HID;
    ao[row + 0]  = (h16)(O0[r] * inv);
    ao[row + 16] = (h16)(O1[r] * inv);
    ao[row + 32] = (h16)(O2[r] * inv);
    ao[row + 48] = (h16)(O3[r] * inv);
  }
}

// ---------------------------------------------------------------------------
// Kernel 4: output projection. out[b,o,i] = sum_c AO[b,i,c]*Wout[o,c] + bout[o]
// A read directly from row-major f16 AO (already in A-friendly order).
// ---------------------------------------------------------------------------
__global__ __launch_bounds__(256) void outproj_kernel(const h16* __restrict__ AO,
                                                      const h16* __restrict__ Wout16,
                                                      const float* __restrict__ bout,
                                                      float* __restrict__ out) {
  const int b  = blockIdx.y;
  const int pt = blockIdx.x;                 // 64-pixel tile
  const int tid = threadIdx.x;
  const int wv = tid >> 5, lane = tid & 31;
  const int lane15 = lane & 15, hi = lane >> 4;
  const int base8 = hi * 8, base16 = hi * 16;

  const int mt = wv >> 1;                    // m-tile 0..3
  const int i0 = pt * 64 + mt * 16 + lane15;
  const h16* arow = AO + ((size_t)b * NQ + i0) * HID;

  v16h Aq[16];                               // K=512 -> 16 chained WMMAs
#pragma unroll
  for (int kc = 0; kc < 16; ++kc)
    Aq[kc] = cat8(*(const v8h*)(arow + kc * 32 + base16),
                  *(const v8h*)(arow + kc * 32 + base16 + 8));

  const int nbase = (wv & 1) * 128;
  for (int nt = 0; nt < 8; ++nt) {
    const int o = nbase + nt * 16 + lane15;
    const h16* wrow = Wout16 + (size_t)o * HID + base16;
    v8f acc = {};
#pragma unroll
    for (int kc = 0; kc < 16; ++kc)
      acc = WMMA_F16(Aq[kc], *(const v16h*)(wrow + kc * 32), acc);

    const float bo = bout[o];
    float* od = out + ((size_t)b * 256 + o) * NQ + pt * 64 + mt * 16;
#pragma unroll
    for (int r = 0; r < 8; ++r)
      od[r + 8 * hi] = acc[r] + bo;
  }
}

// ---------------------------------------------------------------------------
extern "C" void kernel_launch(void* const* d_in, const int* in_sizes, int n_in,
                              void* d_out, int out_size, void* d_ws, size_t ws_size,
                              hipStream_t stream) {
  const float* x    = (const float*)d_in[0];
  const float* a    = (const float*)d_in[1];
  const float* Wq   = (const float*)d_in[2];
  const float* Wkv  = (const float*)d_in[3];
  const float* Wout = (const float*)d_in[4];
  const float* bout = (const float*)d_in[5];
  float* out = (float*)d_out;

  // workspace layout (f16): Q | K | V^T | attn-out | Wq16 | Wkv16 | Wout16
  h16* Q    = (h16*)d_ws;
  h16* K    = Q    + (size_t)BATCH * HEADS * NQ * DHEAD;
  h16* VT   = K    + (size_t)BATCH * HEADS * NK * DHEAD;
  h16* AO   = VT   + (size_t)BATCH * HEADS * DHEAD * NK;
  h16* Wq16 = AO   + (size_t)BATCH * NQ * HID;
  h16* Wkv16  = Wq16  + (size_t)HID * CIN;
  h16* Wout16 = Wkv16 + (size_t)2 * HID * CIN;

  wcvt_kernel<<<dim3(128), 256, 0, stream>>>(Wq,   Wq16,   HID * CIN);
  wcvt_kernel<<<dim3(256), 256, 0, stream>>>(Wkv,  Wkv16,  2 * HID * CIN);
  wcvt_kernel<<<dim3(128), 256, 0, stream>>>(Wout, Wout16, CIN * HID);

  qproj_kernel  <<<dim3(64, 4),    256, 0, stream>>>(x, Wq16, Q);
  kvproj_kernel <<<dim3(16, 2, 4), 256, 0, stream>>>(a, Wkv16, K, VT);
  flash_kernel  <<<dim3(2048),     128, 0, stream>>>(Q, K, VT, AO);
  outproj_kernel<<<dim3(64, 4),    256, 0, stream>>>(AO, Wout16, bout, out);
}
